// GCNPolicy_51986284150875
// MI455X (gfx1250) — matrix operand order repaired and checked by reading, not compiled, not run on hardware
//
#include <hip/hip_runtime.h>
#include <cstdint>
#include <cstddef>

// ---------------------------------------------------------------------------
// GCN policy net for MI455X (gfx1250, wave32, WMMA).
// B=8, N=4096, F_IN=3, H_GNN=OUT_GNN=3, HID=128, N_ACT=16, IN_DIM=16420.
//
// Roofline: dominated by one streaming read of adj (536 MB ~ 23 us @ 23.3 TB/s).
// adj is read ONCE (nontemporal), bit-packed to 16.8 MB (L2-resident for both
// GCN layers). Aggregation = dinv_i * sum_j A_ij * (dinv_j * feat_j) is done as
// aggT = sT x AT with v_wmma_f32_16x16x32_bf16: A bits are exact {0,1} in bf16,
// features bf16, accumulation f32 over 4096 terms.
//
//   1) pack_kernel : adj -> packed bits of (adj!=0 | I) + dinv = rsqrt(deg)
//   2) s1_kernel   : s1T[b][f][j] = dinv_j * (x_j @ W1)[f]  (bf16, row 3 = 0)
//   3) agg<1>      : WMMA; epilogue relu(+b1), @W2, *dinv -> s2T (row 3 = 0)
//   4) agg<2>      : WMMA; epilogue +b2 -> h2 (f32)
//   5) fc_kernel   : z=[idx|h2|y] -> relu(z@fcW1+fcb1) @ fcW2 + fcb2
// ---------------------------------------------------------------------------

#define BB    8
#define NN    4096
#define NWRD  128          // NN/32 packed words per adjacency row
#define HID   128
#define NACT  16
#define INDIM 16420        // NN + 3*NN + 36

typedef __attribute__((ext_vector_type(16))) __bf16 v16bf;
typedef __attribute__((ext_vector_type(8)))  float  v8f;

union Frag16 { unsigned u[8]; v16bf v; };

// ---------------------------------------------------------------------------
// Pass 1: bit-pack (adj!=0 | I) and compute dinv. One wave per row.
// ---------------------------------------------------------------------------
__global__ __launch_bounds__(256) void pack_kernel(const int* __restrict__ adj,
                                                   unsigned* __restrict__ pA,
                                                   float* __restrict__ dinv) {
  const int wid  = (blockIdx.x * blockDim.x + threadIdx.x) >> 5;  // row id in [0, B*N)
  const int lane = threadIdx.x & 31;
  const int i    = wid & (NN - 1);
  const int* row = adj + (size_t)wid * NN;
  unsigned* prow = pA + (size_t)wid * NWRD;
  int cnt = 0;
  for (int w = 0; w < NWRD; ++w) {
    const int j = (w << 5) + lane;
    const int v = __builtin_nontemporal_load(row + j);   // stream, don't pollute L2
    const bool bit = (v != 0) || (j == i);               // self-loop forced to 1
    const unsigned mask = __builtin_amdgcn_ballot_w32(bit);
    if (lane == 0) { prow[w] = mask; cnt += __popc(mask); }
  }
  if (lane == 0) dinv[wid] = rsqrtf((float)cnt);
}

// ---------------------------------------------------------------------------
// Pass 2: s1T[b][f][j] = dinv_j * (x_j @ W1)[f], bf16 transposed; row f=3 -> 0
// (padding row read by the non-feature lanes of the WMMA A-operand).
// ---------------------------------------------------------------------------
__global__ __launch_bounds__(256) void s1_kernel(const float* __restrict__ x,
                                                 const float* __restrict__ dinv,
                                                 const float* __restrict__ W1,
                                                 __bf16* __restrict__ s1T) {
  const int id = blockIdx.x * blockDim.x + threadIdx.x;  // b*N + j
  const int b  = id >> 12;
  const int j  = id & (NN - 1);
  const float x0 = x[3 * id + 0], x1 = x[3 * id + 1], x2 = x[3 * id + 2];
  const float d = dinv[id];
#pragma unroll
  for (int f = 0; f < 3; ++f) {
    const float s = d * (x0 * W1[0 * 3 + f] + x1 * W1[1 * 3 + f] + x2 * W1[2 * 3 + f]);
    s1T[((size_t)b * 4 + f) * NN + j] = (__bf16)s;
  }
  s1T[((size_t)b * 4 + 3) * NN + j] = (__bf16)0.0f;      // zero padding row
}

// ---------------------------------------------------------------------------
// Aggregation: aggT[f][i] = sum_j sT[f][j] * A[i][j], via v_wmma_f32_16x16x32_bf16.
// One wave per 16-node output tile; K loop over 4096 j in steps of 32.
//   A-operand  (16x32 bf16): feature rows (0..2 live; lanes 3..15 clamped to the
//                            zeroed padding row 3 -> uniform, branch-free loads)
//   B-operand  (32x16 bf16): adjacency bits expanded to {0,1} bf16 (exact)
//   C/D        (16x16 f32) : lane half 0, acc[0..2] = features 0..2
// Two accumulators (even/odd K-steps) keep two WMMA+load pipes in flight.
// ---------------------------------------------------------------------------
template <int LAYER>
__global__ __launch_bounds__(256) void agg_kernel(const unsigned* __restrict__ pA,
                                                  const __bf16* __restrict__ sT,
                                                  const float* __restrict__ dinv,
                                                  const float* __restrict__ Wnext,
                                                  const float* __restrict__ bias,
                                                  __bf16* __restrict__ sOut,
                                                  float* __restrict__ h2) {
  const int wave  = (blockIdx.x * blockDim.x + threadIdx.x) >> 5;  // 0..2047
  const int lane  = threadIdx.x & 31;
  const int b     = wave >> 8;            // 256 tiles per batch
  const int ibase = (wave & 255) << 4;
  const int half  = lane >> 4;
  const int l15   = lane & 15;
  const int frow  = (l15 < 3) ? l15 : 3;  // non-feature lanes -> zero padding row

  const unsigned* rowA = pA + ((size_t)b * NN + ibase + l15) * NWRD;   // bits of A[i][:]
  const __bf16*  sRow  = sT + ((size_t)b * 4 + frow) * NN + 16 * half;  // feature row

  v8f acc0 = {};
  v8f acc1 = {};
  for (int m = 0; m < NWRD / 4; ++m) {
    const uint4 P = *(const uint4*)(rowA + 4 * m);   // 4 packed words = 128 j bits
#pragma unroll
    for (int q = 0; q < 4; ++q) {
      const int j0 = ((4 * m + q) << 5);
      const unsigned word = (q == 0) ? P.x : (q == 1) ? P.y : (q == 2) ? P.z : P.w;
      const unsigned bits = (word >> (16 * half)) & 0xFFFFu;  // this half's K range

      // B operand: 16 adjacency bits -> 16 bf16 {0,1} (0x3F80 == bf16 1.0)
      Frag16 bf;
#pragma unroll
      for (int p = 0; p < 8; ++p) {
        bf.u[p] = (((bits >> (2 * p)) & 1u) ? 0x00003F80u : 0u) |
                  (((bits >> (2 * p + 1)) & 1u) ? 0x3F800000u : 0u);
      }

      // A operand: sT[f][j0 + 16h .. j0 + 16h+15] (contiguous K, matches B bits)
      Frag16 af;
      {
        const __bf16* p0 = sRow + j0;
        const uint4 c0 = *(const uint4*)p0;
        const uint4 c1 = *(const uint4*)(p0 + 8);
        af.u[0] = c0.x; af.u[1] = c0.y; af.u[2] = c0.z; af.u[3] = c0.w;
        af.u[4] = c1.x; af.u[5] = c1.y; af.u[6] = c1.z; af.u[7] = c1.w;
      }

      if (q & 1) {
        acc1 = __builtin_amdgcn_wmma_f32_16x16x32_bf16(false, af.v, false, bf.v,
                                                       (short)0, acc1, false, false);
      } else {
        acc0 = __builtin_amdgcn_wmma_f32_16x16x32_bf16(false, af.v, false, bf.v,
                                                       (short)0, acc0, false, false);
      }
    }
  }
  const v8f acc = acc0 + acc1;

  if (half == 0) {  // lanes 0..15 hold output rows 0..7 -> features 0..2 in acc[0..2]
    const int i = ibase + l15;
    const float d = dinv[b * NN + i];
    if (LAYER == 1) {
      const float h0 = fmaxf(d * acc[0] + bias[0], 0.0f);
      const float h1 = fmaxf(d * acc[1] + bias[1], 0.0f);
      const float hv = fmaxf(d * acc[2] + bias[2], 0.0f);
#pragma unroll
      for (int g = 0; g < 3; ++g) {
        const float s = d * (h0 * Wnext[0 * 3 + g] + h1 * Wnext[1 * 3 + g] + hv * Wnext[2 * 3 + g]);
        sOut[((size_t)b * 4 + g) * NN + i] = (__bf16)s;
      }
      sOut[((size_t)b * 4 + 3) * NN + i] = (__bf16)0.0f;  // keep padding row zero
    } else {
#pragma unroll
      for (int f = 0; f < 3; ++f)
        h2[((size_t)b * NN + i) * 3 + f] = d * acc[f] + bias[f];
    }
  }
}

// ---------------------------------------------------------------------------
// Head: z = [idx | h2 | y] (16420) -> relu(z@fcW1 + fcb1) -> @fcW2 + fcb2.
// One block per batch row; 4 K-slices x 128 hidden units; LDS reduction.
// ---------------------------------------------------------------------------
__global__ __launch_bounds__(512) void fc_kernel(const float* __restrict__ idx,
                                                 const float* __restrict__ y,
                                                 const float* __restrict__ h2,
                                                 const float* __restrict__ fcW1,
                                                 const float* __restrict__ fcb1,
                                                 const float* __restrict__ fcW2,
                                                 const float* __restrict__ fcb2,
                                                 float* __restrict__ out) {
  const int b = blockIdx.x;
  const int t = threadIdx.x;
  const int hid = t & (HID - 1);
  const int slice = t >> 7;  // 0..3
  __shared__ float part[4][HID];
  __shared__ float z1[HID];

  const int per = (INDIM + 3) / 4;
  const int n0 = slice * per;
  const int n1 = (n0 + per < INDIM) ? (n0 + per) : INDIM;
  float acc = 0.0f;
  for (int n = n0; n < n1; ++n) {
    float z;
    if (n < NN)              z = idx[b * NN + n];
    else if (n < 4 * NN)     z = h2[(size_t)b * 3 * NN + (n - NN)];
    else                     z = y[b * 36 + (n - 4 * NN)];
    acc += z * fcW1[(size_t)n * HID + hid];
  }
  part[slice][hid] = acc;
  __syncthreads();
  if (t < HID) {
    const float v = part[0][t] + part[1][t] + part[2][t] + part[3][t] + fcb1[t];
    z1[t] = fmaxf(v, 0.0f);
  }
  __syncthreads();
  if (t < NACT) {
    float o = fcb2[t];
#pragma unroll 8
    for (int k = 0; k < HID; ++k) o += z1[k] * fcW2[k * NACT + t];
    out[b * NACT + t] = o;
  }
}

// ---------------------------------------------------------------------------
extern "C" void kernel_launch(void* const* d_in, const int* in_sizes, int n_in,
                              void* d_out, int out_size, void* d_ws, size_t ws_size,
                              hipStream_t stream) {
  (void)in_sizes; (void)n_in; (void)out_size; (void)ws_size;

  const float* idx  = (const float*)d_in[0];
  const float* x    = (const float*)d_in[1];
  const float* y    = (const float*)d_in[2];
  const int*   adj  = (const int*)  d_in[3];
  const float* W1   = (const float*)d_in[4];
  const float* b1   = (const float*)d_in[5];
  const float* W2   = (const float*)d_in[6];
  const float* b2   = (const float*)d_in[7];
  const float* fcW1 = (const float*)d_in[8];
  const float* fcb1 = (const float*)d_in[9];
  const float* fcW2 = (const float*)d_in[10];
  const float* fcb2 = (const float*)d_in[11];
  float* out = (float*)d_out;

  // Workspace carve-up (all offsets 256B aligned by construction).
  char* ws = (char*)d_ws;
  unsigned* pA   = (unsigned*)ws;                       // 8*4096*128*4  = 16,777,216 B
  float*    dinv = (float*)(ws + 16777216);             // 8*4096*4     =    131,072 B
  __bf16*   s1T  = (__bf16*)(ws + 16908288);            // 8*4*4096*2   =    262,144 B
  __bf16*   s2T  = (__bf16*)(ws + 17170432);            // 8*4*4096*2   =    262,144 B
  float*    h2   = (float*)(ws + 17432576);             // 8*4096*3*4   =    393,216 B
                                                        // total ~17.8 MB

  // 1) pack + degree: one wave per adjacency row (32768 waves).
  pack_kernel<<<dim3(4096), dim3(256), 0, stream>>>(adj, pA, dinv);

  // 2) scaled input features (+ zero padding row).
  s1_kernel<<<dim3((BB * NN) / 256), dim3(256), 0, stream>>>(x, dinv, W1, s1T);

  // 3) GCN layer 1 aggregation (WMMA) + fused relu/W2/dinv epilogue.
  agg_kernel<1><<<dim3(256), dim3(256), 0, stream>>>(pA, s1T, dinv, W2, b1, s2T, nullptr);

  // 4) GCN layer 2 aggregation (WMMA) -> h2 (f32).
  agg_kernel<2><<<dim3(256), dim3(256), 0, stream>>>(pA, s2T, dinv, nullptr, b2, nullptr, h2);

  // 5) MLP head.
  fc_kernel<<<dim3(BB), dim3(512), 0, stream>>>(idx, y, h2, fcW1, fcb1, fcW2, fcb2, out);
}
